// MultiHeadedSelfAttention_49108656062941
// MI455X (gfx1250) — matrix operand order repaired and checked by reading, not compiled, not run on hardware
//
#include <hip/hip_runtime.h>

// ---------------------------------------------------------------------------
// MHA forward, B=2, S=2048, D=768, H=12, HD=64 on gfx1250 (CDNA5, wave32).
// Phase 0: one-shot f32->bf16 conversion of inputs + weights (memory bound).
// Phase 1: bf16 WMMA projection GEMMs (X @ W^T + b), per-head bf16 layouts.
// Phase 2: flash attention, 64-key blocks, bf16 WMMA, f32 accumulate.
// ---------------------------------------------------------------------------

#define B_  2
#define S_  2048
#define D_  768
#define H_  12
#define HD_ 64

typedef __attribute__((ext_vector_type(16))) unsigned short v16u;
typedef __attribute__((ext_vector_type(8)))  unsigned short v8u;
typedef __attribute__((ext_vector_type(16))) __bf16         v16bf;
typedef __attribute__((ext_vector_type(8)))  float          v8f;

static __device__ __forceinline__ unsigned short f2bf(float f) {
  // round-to-nearest-even f32 -> bf16
  unsigned u = __builtin_bit_cast(unsigned, f);
  u += 0x7fffu + ((u >> 16) & 1u);
  return (unsigned short)(u >> 16);
}

static __device__ __forceinline__ unsigned pack2(float lo, float hi) {
  return (unsigned)f2bf(lo) | ((unsigned)f2bf(hi) << 16);
}

static __device__ __forceinline__ v16bf asbf(v16u x) {
  return __builtin_bit_cast(v16bf, x);
}

static __device__ __forceinline__ v16u cat16(v8u a, v8u b) {
  return __builtin_shufflevector(a, b, 0, 1, 2, 3, 4, 5, 6, 7,
                                       8, 9, 10, 11, 12, 13, 14, 15);
}

static __device__ __forceinline__ v8f wmma_bf16(v16u a, v16u b, v8f c) {
  return __builtin_amdgcn_wmma_f32_16x16x32_bf16(
      /*neg_a=*/false, asbf(a), /*neg_b=*/false, asbf(b),
      /*c_mod=*/(short)0, c, /*reuse_a=*/false, /*reuse_b=*/false);
}

static __device__ __forceinline__ v8f vzero8() {
  v8f z = {0.f, 0.f, 0.f, 0.f, 0.f, 0.f, 0.f, 0.f};
  return z;
}

// ---------------------------------------------------------------------------
// Phase 0: elementwise f32 -> bf16, 8 elements/thread, vectorized b128 I/O.
// n8 = element_count / 8 (all tensor sizes here are divisible by 8).
// ---------------------------------------------------------------------------
__global__ void __launch_bounds__(256)
cvt_bf16_kernel(const float* __restrict__ in, unsigned short* __restrict__ out,
                int n8) {
  const int i = blockIdx.x * blockDim.x + threadIdx.x;
  if (i >= n8) return;
  const float4* p = (const float4*)(in + (size_t)i * 8);
  float4 a = p[0], b = p[1];
  uint4 o;
  o.x = pack2(a.x, a.y);
  o.y = pack2(a.z, a.w);
  o.z = pack2(b.x, b.y);
  o.w = pack2(b.z, b.w);
  *(uint4*)(out + (size_t)i * 8) = o;
}

// ---------------------------------------------------------------------------
// Phase 1: Y = X @ W^T + bias, bf16 in (pre-converted), bf16 out.
// grid = (S/128, D/64, B), block = 128 (4 waves).
// Wave w computes rows [blk.x*128 + w*32, +32) x cols [blk.y*64, +64):
//   2 A-fragments x 4 B-fragments -> 8 WMMAs / K-step, B reused across rows.
// vmode==0: out[(b*H+h)*S + s][hd]   (Q, K)
// vmode==1: out[(b*H+h)*HD + hd][s]  (V transposed)
// ---------------------------------------------------------------------------
__global__ void __launch_bounds__(128)
mha_proj_kernel(const unsigned short* __restrict__ x,
                const unsigned short* __restrict__ w,
                const float* __restrict__ bias, unsigned short* __restrict__ out,
                int vmode, float scale) {
  const int lane = threadIdx.x & 31;
  const int wave = threadIdx.x >> 5;
  const int half = lane >> 4;   // 0 or 1
  const int l16  = lane & 15;
  const int b    = blockIdx.z;
  const int row0 = blockIdx.x * 128 + wave * 32;
  const int n0   = blockIdx.y * 64;

  const unsigned short* xr0 = x + (size_t)b * S_ * D_ + (size_t)(row0 + l16) * D_;
  const unsigned short* xr1 = xr0 + (size_t)16 * D_;

  v8f acc[2][4] = {{vzero8(), vzero8(), vzero8(), vzero8()},
                   {vzero8(), vzero8(), vzero8(), vzero8()}};

  for (int k0 = 0; k0 < D_; k0 += 32) {
    // A fragments: lane = row, two 8-element chunks per the 16-bit A layout
    v16u a0 = cat16(*(const v8u*)(xr0 + k0 + 8 * half),
                    *(const v8u*)(xr0 + k0 + 16 + 8 * half));
    v16u a1 = cat16(*(const v8u*)(xr1 + k0 + 8 * half),
                    *(const v8u*)(xr1 + k0 + 16 + 8 * half));
#pragma unroll
    for (int t = 0; t < 4; ++t) {
      // B[k][n] = W[n][k]: lane = col n, two 8-element k chunks, same k order as A
      const unsigned short* wr =
          w + (size_t)(n0 + t * 16 + l16) * D_ + k0 + 8 * half;
      v16u bv = cat16(*(const v8u*)wr, *(const v8u*)(wr + 16));
      acc[0][t] = wmma_bf16(a0, bv, acc[0][t]);
      acc[1][t] = wmma_bf16(a1, bv, acc[1][t]);
    }
  }

  // epilogue: bias, scale, scatter to per-head bf16 layout
#pragma unroll
  for (int rt = 0; rt < 2; ++rt) {
    const int rowb = row0 + rt * 16;
#pragma unroll
    for (int t = 0; t < 4; ++t) {
      const int   o    = n0 + t * 16 + l16;
      const float bv_  = bias[o];
      const int   hIdx = o >> 6;
      const int   hd   = o & 63;
#pragma unroll
      for (int r = 0; r < 8; ++r) {
        const int   m = rowb + r + 8 * half;   // C-layout row
        const float v = (acc[rt][t][r] + bv_) * scale;
        size_t idx;
        if (vmode)
          idx = ((size_t)(b * H_ + hIdx) * HD_ + hd) * S_ + m;
        else
          idx = ((size_t)(b * H_ + hIdx) * S_ + m) * HD_ + hd;
        out[idx] = f2bf(v);
      }
    }
  }
}

// ---------------------------------------------------------------------------
// Phase 2: flash attention.  grid = (S/64, H, B), block = 128 (4 waves).
// Each wave owns a 16-query tile; online softmax over 64-key blocks:
// 8 score WMMAs + 8 PV WMMAs per iteration.
// Q,K: bf16 [B][H][S][HD]; V: bf16 [B][H][HD][S]; out: fp32 (B,S,D).
// ---------------------------------------------------------------------------
__global__ void __launch_bounds__(128)
mha_attn_kernel(const unsigned short* __restrict__ Qp,
                const unsigned short* __restrict__ Kp,
                const unsigned short* __restrict__ Vt,
                float* __restrict__ out) {
  __shared__ __attribute__((aligned(16))) unsigned short plds[4 * 16 * 64];

  const int lane = threadIdx.x & 31;
  const int wave = threadIdx.x >> 5;
  const int half = lane >> 4;
  const int l16  = lane & 15;
  const int b    = blockIdx.z;
  const int h    = blockIdx.y;
  const int q0   = blockIdx.x * 64 + wave * 16;

  const unsigned short* Qh = Qp + (size_t)(b * H_ + h) * S_ * HD_;
  const unsigned short* Kh = Kp + (size_t)(b * H_ + h) * S_ * HD_;
  const unsigned short* Vh = Vt + (size_t)(b * H_ + h) * HD_ * S_;
  unsigned short*       pl = plds + wave * (16 * 64);

  // Q A-fragments (16 x 64 = two 16x32 fragments), loaded once
  v16u qa0, qa1;
  {
    const unsigned short* qrow = Qh + (size_t)(q0 + l16) * HD_;
    qa0 = cat16(*(const v8u*)(qrow + 8 * half),
                *(const v8u*)(qrow + 16 + 8 * half));
    qa1 = cat16(*(const v8u*)(qrow + 32 + 8 * half),
                *(const v8u*)(qrow + 48 + 8 * half));
  }

  float mrow[8], lrow[8];
#pragma unroll
  for (int r = 0; r < 8; ++r) { mrow[r] = -3.0e38f; lrow[r] = 0.f; }
  v8f acc[4] = {vzero8(), vzero8(), vzero8(), vzero8()};

  for (int j0 = 0; j0 < S_; j0 += 64) {
    if (j0 + 64 < S_) {  // uniform: EXEC stays all-ones
      __builtin_prefetch(Kh + (size_t)(j0 + 64) * HD_, 0, 3);
      __builtin_prefetch(Vh + (j0 + 64), 0, 3);
    }

    // ---- scores: four 16-key tiles, K-dim = HD = 64 (2 WMMAs each) ----
    v8f s[4] = {vzero8(), vzero8(), vzero8(), vzero8()};
#pragma unroll
    for (int t = 0; t < 4; ++t) {
      const unsigned short* kr = Kh + (size_t)(j0 + t * 16 + l16) * HD_ + 8 * half;
      v16u kb0 = cat16(*(const v8u*)kr, *(const v8u*)(kr + 16));         // hd 0..31
      v16u kb1 = cat16(*(const v8u*)(kr + 32), *(const v8u*)(kr + 48)); // hd 32..63
      s[t] = wmma_bf16(qa0, kb0, s[t]);
      s[t] = wmma_bf16(qa1, kb1, s[t]);
    }

    // ---- online softmax over this 64-key block ----
    float sc[8];
#pragma unroll
    for (int r = 0; r < 8; ++r) {
      float bm = fmaxf(fmaxf(s[0][r], s[1][r]), fmaxf(s[2][r], s[3][r]));
      bm = fmaxf(bm, __shfl_xor(bm, 1));
      bm = fmaxf(bm, __shfl_xor(bm, 2));
      bm = fmaxf(bm, __shfl_xor(bm, 4));
      bm = fmaxf(bm, __shfl_xor(bm, 8));   // row max across the 16 key lanes
      const float mnew = fmaxf(mrow[r], bm);
      const float scl  = __expf(mrow[r] - mnew);
      const float p0   = __expf(s[0][r] - mnew);
      const float p1   = __expf(s[1][r] - mnew);
      const float p2   = __expf(s[2][r] - mnew);
      const float p3   = __expf(s[3][r] - mnew);
      lrow[r] = lrow[r] * scl + ((p0 + p1) + (p2 + p3));
      mrow[r] = mnew;
      sc[r]   = scl;
      const int rr = r + 8 * half;          // C-layout row
      pl[rr * 64 + l16]      = f2bf(p0);
      pl[rr * 64 + 16 + l16] = f2bf(p1);
      pl[rr * 64 + 32 + l16] = f2bf(p2);
      pl[rr * 64 + 48 + l16] = f2bf(p3);
    }
#pragma unroll
    for (int t = 0; t < 4; ++t)
#pragma unroll
      for (int r = 0; r < 8; ++r) acc[t][r] *= sc[r];

    // private per-wave LDS slice: same-wave RAW, dscnt wait is enough
    asm volatile("s_wait_dscnt 0" ::: "memory");

    // ---- P A-fragments (16 queries x 64 keys) from LDS ----
    v16u pa0, pa1;
    {
      const unsigned short* prow = pl + l16 * 64;
      pa0 = cat16(*(const v8u*)(prow + 8 * half),
                  *(const v8u*)(prow + 16 + 8 * half));
      pa1 = cat16(*(const v8u*)(prow + 32 + 8 * half),
                  *(const v8u*)(prow + 48 + 8 * half));
    }

    // ---- O += P x V: V^T rows contiguous along keys ----
#pragma unroll
    for (int t = 0; t < 4; ++t) {
      const unsigned short* vr = Vh + (size_t)(t * 16 + l16) * S_ + j0 + 8 * half;
      v16u vb0 = cat16(*(const v8u*)vr, *(const v8u*)(vr + 16));         // keys 0..31
      v16u vb1 = cat16(*(const v8u*)(vr + 32), *(const v8u*)(vr + 48)); // keys 32..63
      acc[t] = wmma_bf16(pa0, vb0, acc[t]);
      acc[t] = wmma_bf16(pa1, vb1, acc[t]);
    }
  }

  // ---- finalize: divide by row sums, write fp32 (B,S,D) ----
  float inv[8];
#pragma unroll
  for (int r = 0; r < 8; ++r) {
    float l = lrow[r];
    l += __shfl_xor(l, 1);
    l += __shfl_xor(l, 2);
    l += __shfl_xor(l, 4);
    l += __shfl_xor(l, 8);
    inv[r] = 1.0f / l;
  }
#pragma unroll
  for (int t = 0; t < 4; ++t) {
    const int col = h * 64 + t * 16 + l16;
#pragma unroll
    for (int r = 0; r < 8; ++r) {
      const int m = q0 + r + 8 * half;
      out[((size_t)b * S_ + m) * (size_t)D_ + col] = acc[t][r] * inv[r];
    }
  }
}

// ---------------------------------------------------------------------------
// Launch: 6 bf16 conversions, 3 projection GEMMs, flash attention.
// Workspace (bf16): xq|xk|xv (3x 6.3MB) + Wq|Wk|Wv (3x 1.2MB)
//                 + Qp|Kp|Vt (3x 6.3MB)  => ~41.3 MB.
// ---------------------------------------------------------------------------
extern "C" void kernel_launch(void* const* d_in, const int* in_sizes, int n_in,
                              void* d_out, int out_size, void* d_ws, size_t ws_size,
                              hipStream_t stream) {
  (void)in_sizes; (void)n_in; (void)out_size; (void)ws_size;

  const float* q  = (const float*)d_in[0];
  const float* k  = (const float*)d_in[1];
  const float* v  = (const float*)d_in[2];
  const float* Wq = (const float*)d_in[3];
  const float* bq = (const float*)d_in[4];
  const float* Wk = (const float*)d_in[5];
  const float* bk = (const float*)d_in[6];
  const float* Wv = (const float*)d_in[7];
  const float* bv = (const float*)d_in[8];
  float*       o  = (float*)d_out;

  const size_t nx = (size_t)B_ * S_ * D_;   // 3,145,728
  const size_t nw = (size_t)D_ * D_;        //   589,824

  unsigned short* ws  = (unsigned short*)d_ws;
  unsigned short* xq  = ws;
  unsigned short* xk  = xq + nx;
  unsigned short* xv  = xk + nx;
  unsigned short* wqb = xv + nx;
  unsigned short* wkb = wqb + nw;
  unsigned short* wvb = wkb + nw;
  unsigned short* qp  = wvb + nw;
  unsigned short* kp  = qp + nx;
  unsigned short* vt  = kp + nx;

  const int nx8 = (int)(nx / 8), nw8 = (int)(nw / 8);
  const dim3 cblk(256);
  cvt_bf16_kernel<<<dim3((nx8 + 255) / 256), cblk, 0, stream>>>(q, xq, nx8);
  cvt_bf16_kernel<<<dim3((nx8 + 255) / 256), cblk, 0, stream>>>(k, xk, nx8);
  cvt_bf16_kernel<<<dim3((nx8 + 255) / 256), cblk, 0, stream>>>(v, xv, nx8);
  cvt_bf16_kernel<<<dim3((nw8 + 255) / 256), cblk, 0, stream>>>(Wq, wqb, nw8);
  cvt_bf16_kernel<<<dim3((nw8 + 255) / 256), cblk, 0, stream>>>(Wk, wkb, nw8);
  cvt_bf16_kernel<<<dim3((nw8 + 255) / 256), cblk, 0, stream>>>(Wv, wvb, nw8);

  const dim3 pgrid(S_ / 128, D_ / 64, B_);
  const dim3 agrid(S_ / 64, H_, B_);
  const dim3 blk(128);
  const float qscale = 0.125f;  // 1/sqrt(HD), folded into Q projection

  mha_proj_kernel<<<pgrid, blk, 0, stream>>>(xq, wqb, bq, qp, 0, qscale);
  mha_proj_kernel<<<pgrid, blk, 0, stream>>>(xk, wkb, bk, kp, 0, 1.0f);
  mha_proj_kernel<<<pgrid, blk, 0, stream>>>(xv, wvb, bv, vt, 1, 1.0f);
  mha_attn_kernel<<<agrid, blk, 0, stream>>>(qp, kp, vt, o);
}
